// CharModel_17978733101188
// MI455X (gfx1250) — matrix operand order repaired and checked by reading, not compiled, not run on hardware
//
#include <hip/hip_runtime.h>
#include <hip/hip_bf16.h>

#define SEQ    4096
#define DMODEL 1024
#define NVOCAB 50257

typedef __attribute__((ext_vector_type(16))) __bf16 v16bf;
typedef __attribute__((ext_vector_type(8)))  __bf16 v8bf;
typedef __attribute__((ext_vector_type(8)))  float  v8f;
typedef __attribute__((ext_vector_type(4)))  int    v4i;

typedef __attribute__((address_space(1))) v4i as1_v4i;  // global
typedef __attribute__((address_space(3))) v4i as3_v4i;  // LDS

#define LDS_LD 40  // 32 K-elements + 16B pad per row (80B row stride; breaks 64-bank periodicity)

#if defined(__has_builtin)
#if __has_builtin(__builtin_amdgcn_global_load_async_to_lds_b128)
#define HAVE_ASYNC_LDS 1
#endif
#endif

static __device__ __forceinline__ v16bf combine16(v8bf lo, v8bf hi) {
  return __builtin_shufflevector(lo, hi, 0,1,2,3,4,5,6,7,8,9,10,11,12,13,14,15);
}

static __device__ __forceinline__ v8f wmma_bf16(v16bf a, v16bf b, v8f c) {
  // D = A(16x32 bf16) * B(32x16 bf16) + C(16x16 f32)
  return __builtin_amdgcn_wmma_f32_16x16x32_bf16(false, a, false, b, (short)0, c, false, false);
}

// A fragment: 16x32 bf16, row-major source, K contiguous.
// lane<16 holds M=lane, K {0..7, 16..23}; lane>=16: K {8..15, 24..31}
static __device__ __forceinline__ v16bf load_a_frag(const __bf16* base, int ld, int m0) {
  const int lane = threadIdx.x & 31;
  const int r = lane & 15, h = lane >> 4;
  const __bf16* p = base + (m0 + r) * ld + h * 8;
  return combine16(*(const v8bf*)p, *(const v8bf*)(p + 16));
}

// B fragment: 32x16 bf16 from NT storage ([Ncols][K] row-major, K contiguous).
// lane<16 holds N=lane, K {0..7, 16..23}; lane>=16 holds K {8..15, 24..31}
static __device__ __forceinline__ v16bf load_bt_frag(const __bf16* base, int ld, int n0) {
  const int lane = threadIdx.x & 31;
  const int c = lane & 15, h = lane >> 4;
  const __bf16* p = base + (n0 + c) * ld + h * 8;
  return combine16(*(const v8bf*)p, *(const v8bf*)(p + 16));
}

// 16-byte global -> LDS copy (async path when the builtin is available)
static __device__ __forceinline__ void copy16(const __bf16* g, __bf16* l) {
#ifdef HAVE_ASYNC_LDS
  // AS-qualified pointers must be produced via integer round-trip (no direct
  // reinterpret_cast across address spaces). Low 32 bits of a flat shared
  // pointer are the LDS byte offset.
  as1_v4i* gp = (as1_v4i*)(unsigned long long)(size_t)g;
  as3_v4i* lp = (as3_v4i*)(unsigned)(size_t)l;
  __builtin_amdgcn_global_load_async_to_lds_b128(gp, lp, 0, 0);
#else
  *(v8bf*)l = *(const v8bf*)g;
#endif
}

static __device__ __forceinline__ void async_join() {
#ifdef HAVE_ASYNC_LDS
#if __has_builtin(__builtin_amdgcn_s_wait_asynccnt)
  __builtin_amdgcn_s_wait_asynccnt(0);
#else
  asm volatile("s_wait_asynccnt 0x0" ::: "memory");
#endif
#endif
}

// Stage a [ROWS][32] bf16 tile (NT operand, K contiguous, ld=K) into LDS [ROWS][LDS_LD].
// Row indices clamped to rowmax (vocab tail: duplicates row data; those columns are
// discarded by the guarded store).
template <int ROWS>
static __device__ __forceinline__ void stage_nt(const __bf16* __restrict__ g, int ld,
                                                int row0, int rowmax, int k0,
                                                __bf16* __restrict__ s) {
  constexpr int ITER = (ROWS * 4) / 256;  // 16B chunks per thread
  const int t = threadIdx.x;
#pragma unroll
  for (int it = 0; it < ITER; ++it) {
    const int c = t + it * 256;
    const int row = c >> 2, ko = (c & 3) * 8;
    int gr = row0 + row;
    gr = (gr > rowmax) ? rowmax : gr;
    copy16(g + (size_t)gr * ld + k0 + ko, s + row * LDS_LD + ko);
  }
}

// ------------------------------------------------------------- NT GEMM -----
// C[m,n] = scale * sum_k A[m,k]*B[n,k] (+ bias[n])
// Block tile 128 x BN (double-buffered LDS), 8 waves as 2x4, wave tile 64 x BN/4.
// MODE: 0 = plain, 1 = causal block-skip (scores), 2 = causal K-clip (A@V).
template <int BN, int OUTF32, int BIAS, int MODE>
__global__ __launch_bounds__(256) void k_gemm_nt(const __bf16* __restrict__ Am,
                                                 const __bf16* __restrict__ Bm,
                                                 const float* __restrict__ bias,
                                                 void* __restrict__ Cout,
                                                 int N, int K, int ldc, float scale) {
  constexpr int BM = 128;
  constexpr int BNW = BN / 4;   // wave tile N
  constexpr int NB = BNW / 16;  // B fragments per wave

  const int blk_n = blockIdx.x * BN;
  const int blk_m = blockIdx.y * BM;

  if (MODE == 1 && blk_n > blk_m + BM - 1) return;  // fully above causal diagonal

  int kend = K;
  if (MODE == 2) kend = (blk_m + BM < K) ? (blk_m + BM) : K;  // A[m,k]==0 for k>m
  const int nk = kend >> 5;

  __shared__ __bf16 sA[2][BM * LDS_LD];
  __shared__ __bf16 sB[2][BN * LDS_LD];

  const int lane = threadIdx.x & 31, w = threadIdx.x >> 5;
  const int mw = (w >> 2) * 64;   // wave M offset within block
  const int nw = (w & 3) * BNW;   // wave N offset within block

  v8f acc[4][NB] = {};

  stage_nt<BM>(Am, K, blk_m, 0x7fffffff, 0, sA[0]);
  stage_nt<BN>(Bm, K, blk_n, N - 1, 0, sB[0]);
  async_join();
  __syncthreads();

  for (int kb = 0; kb < nk; ++kb) {
    const int cur = kb & 1;
    if (kb + 1 < nk) {  // prefetch next K-slab into the other buffer
      stage_nt<BM>(Am, K, blk_m, 0x7fffffff, (kb + 1) * 32, sA[cur ^ 1]);
      stage_nt<BN>(Bm, K, blk_n, N - 1, (kb + 1) * 32, sB[cur ^ 1]);
    }
    v16bf bf[NB];
#pragma unroll
    for (int tn = 0; tn < NB; ++tn) bf[tn] = load_bt_frag(sB[cur], LDS_LD, nw + tn * 16);
#pragma unroll
    for (int tm = 0; tm < 4; ++tm) {
      v16bf a = load_a_frag(sA[cur], LDS_LD, mw + tm * 16);
#pragma unroll
      for (int tn = 0; tn < NB; ++tn) acc[tm][tn] = wmma_bf16(a, bf[tn], acc[tm][tn]);
    }
    async_join();
    __syncthreads();
  }

  const int col = lane & 15, rb = (lane >> 4) * 8;
#pragma unroll
  for (int tm = 0; tm < 4; ++tm)
#pragma unroll
    for (int tn = 0; tn < NB; ++tn) {
      const int cc = blk_n + nw + tn * 16 + col;
      if (cc >= N) continue;
      const int rr = blk_m + mw + tm * 16 + rb;
      const float bv = BIAS ? bias[cc] : 0.0f;
      v8f c = acc[tm][tn];
      if (OUTF32) {
        float* dst = (float*)Cout + (size_t)rr * ldc + cc;
#pragma unroll
        for (int r = 0; r < 8; ++r) dst[(size_t)r * ldc] = c[r] * scale + bv;
      } else {
        __bf16* dst = (__bf16*)Cout + (size_t)rr * ldc + cc;
#pragma unroll
        for (int r = 0; r < 8; ++r) dst[(size_t)r * ldc] = (__bf16)(c[r] * scale + bv);
      }
    }
}

// ---------------------------------------------------------------- helpers ---

__global__ __launch_bounds__(256) void k_cast_bf16(const float* __restrict__ src,
                                                   __bf16* __restrict__ dst, size_t n) {
  size_t i = (size_t)blockIdx.x * blockDim.x + threadIdx.x;
  const size_t stride = (size_t)gridDim.x * blockDim.x;
  for (; i < n; i += stride) dst[i] = (__bf16)src[i];
}

__global__ __launch_bounds__(256) void k_gather_cast(const int* __restrict__ tokens,
                                                     const float* __restrict__ E,
                                                     __bf16* __restrict__ xb) {
  const int row = blockIdx.x;
  const float* src = E + (size_t)tokens[row] * DMODEL;
  __bf16* dst = xb + (size_t)row * DMODEL;
  for (int j = threadIdx.x; j < DMODEL; j += blockDim.x) dst[j] = (__bf16)src[j];
}

// 32x32-tiled bf16 transpose: dst[c][r] = src[r][c] (src R x C)
__global__ __launch_bounds__(256) void k_transpose(const __bf16* __restrict__ src,
                                                   __bf16* __restrict__ dst, int R, int C) {
  __shared__ __bf16 tile[32][33];
  const int c0 = blockIdx.x * 32, r0 = blockIdx.y * 32;
  for (int i = threadIdx.y; i < 32; i += 8)
    tile[i][threadIdx.x] = src[(size_t)(r0 + i) * C + c0 + threadIdx.x];
  __syncthreads();
  for (int i = threadIdx.y; i < 32; i += 8)
    dst[(size_t)(c0 + i) * R + r0 + threadIdx.x] = tile[threadIdx.x][i];
}

// ------------------------------------------------------- causal softmax ----
// One block per row i: A[i,j] = softmax over j<=i (bf16), zeros for j>i.
__global__ __launch_bounds__(256) void k_softmax(const float* __restrict__ S,
                                                 __bf16* __restrict__ A) {
  const int i = blockIdx.x;
  const int len = i + 1;
  const float* row = S + (size_t)i * SEQ;
  __shared__ float red[256];

  float m = -3.4e38f;
  for (int j = threadIdx.x; j < len; j += 256) m = fmaxf(m, row[j]);
  red[threadIdx.x] = m;
  __syncthreads();
  for (int s = 128; s > 0; s >>= 1) {
    if (threadIdx.x < s) red[threadIdx.x] = fmaxf(red[threadIdx.x], red[threadIdx.x + s]);
    __syncthreads();
  }
  m = red[0];
  __syncthreads();

  float sum = 0.f;
  for (int j = threadIdx.x; j < len; j += 256) sum += __expf(row[j] - m);
  red[threadIdx.x] = sum;
  __syncthreads();
  for (int s = 128; s > 0; s >>= 1) {
    if (threadIdx.x < s) red[threadIdx.x] += red[threadIdx.x + s];
    __syncthreads();
  }
  const float inv = 1.0f / red[0];

  __bf16* arow = A + (size_t)i * SEQ;
  for (int j = threadIdx.x; j < SEQ; j += 256)
    arow[j] = (__bf16)((j < len) ? __expf(row[j] - m) * inv : 0.0f);
}

// ------------------------------------------------------------------ host ---

extern "C" void kernel_launch(void* const* d_in, const int* in_sizes, int n_in,
                              void* d_out, int out_size, void* d_ws, size_t ws_size,
                              hipStream_t stream) {
  const int*   tokens = (const int*)  d_in[0];
  const float* E      = (const float*)d_in[1];
  const float* Wq     = (const float*)d_in[2];
  const float* bq     = (const float*)d_in[3];
  const float* Wk     = (const float*)d_in[4];
  const float* bk     = (const float*)d_in[5];
  const float* Wv     = (const float*)d_in[6];
  const float* bv     = (const float*)d_in[7];
  const float* Wp     = (const float*)d_in[8];
  const float* bp     = (const float*)d_in[9];
  float* logits = (float*)d_out;

  // scratch layout, 256B aligned
  char* p = (char*)d_ws;
  auto carve = [&](size_t bytes) -> void* {
    void* r = (void*)p;
    p += (bytes + 255) & ~(size_t)255;
    return r;
  };
  __bf16* xb   = (__bf16*)carve((size_t)SEQ * DMODEL * 2);
  __bf16* wqb  = (__bf16*)carve((size_t)DMODEL * DMODEL * 2);
  __bf16* wkb  = (__bf16*)carve((size_t)DMODEL * DMODEL * 2);
  __bf16* wvb  = (__bf16*)carve((size_t)DMODEL * DMODEL * 2);
  __bf16* Qb   = (__bf16*)carve((size_t)SEQ * DMODEL * 2);
  __bf16* Kb   = (__bf16*)carve((size_t)SEQ * DMODEL * 2);
  __bf16* Vb   = (__bf16*)carve((size_t)SEQ * DMODEL * 2);
  __bf16* Vt   = (__bf16*)carve((size_t)DMODEL * SEQ * 2);
  float*  S    = (float*) carve((size_t)SEQ * SEQ * 4);
  __bf16* Ab   = (__bf16*)carve((size_t)SEQ * SEQ * 2);
  __bf16* outb = (__bf16*)carve((size_t)SEQ * DMODEL * 2);
  __bf16* wpb  = (__bf16*)carve((size_t)NVOCAB * DMODEL * 2);
  (void)ws_size; (void)in_sizes; (void)n_in; (void)out_size;

  // 1. precision casts + embedding gather
  k_cast_bf16<<<2048, 256, 0, stream>>>(Wq, wqb, (size_t)DMODEL * DMODEL);
  k_cast_bf16<<<2048, 256, 0, stream>>>(Wk, wkb, (size_t)DMODEL * DMODEL);
  k_cast_bf16<<<2048, 256, 0, stream>>>(Wv, wvb, (size_t)DMODEL * DMODEL);
  k_cast_bf16<<<2048, 256, 0, stream>>>(Wp, wpb, (size_t)NVOCAB * DMODEL);
  k_gather_cast<<<SEQ, 256, 0, stream>>>(tokens, E, xb);

  // 2. Q/K/V projections (bf16 out, bias fused)
  dim3 gQKV(DMODEL / 128, SEQ / 128);
  k_gemm_nt<128, 0, 1, 0><<<gQKV, 256, 0, stream>>>(xb, wqb, bq, Qb, DMODEL, DMODEL, DMODEL, 1.0f);
  k_gemm_nt<128, 0, 1, 0><<<gQKV, 256, 0, stream>>>(xb, wkb, bk, Kb, DMODEL, DMODEL, DMODEL, 1.0f);
  k_gemm_nt<128, 0, 1, 0><<<gQKV, 256, 0, stream>>>(xb, wvb, bv, Vb, DMODEL, DMODEL, DMODEL, 1.0f);

  // 3. causal scores (scale fused, upper-triangular blocks skipped) + softmax
  dim3 gS(SEQ / 256, SEQ / 128);
  k_gemm_nt<256, 1, 0, 1><<<gS, 256, 0, stream>>>(Qb, Kb, nullptr, S, SEQ, DMODEL, SEQ, 0.03125f);
  k_softmax<<<SEQ, 256, 0, stream>>>(S, Ab);

  // 4. attention output: transpose V once, then NT GEMM with causal K-clip
  k_transpose<<<dim3(DMODEL / 32, SEQ / 32), dim3(32, 8), 0, stream>>>(Vb, Vt, SEQ, DMODEL);
  dim3 gAV(DMODEL / 128, SEQ / 128);
  k_gemm_nt<128, 0, 0, 2><<<gAV, 256, 0, stream>>>(Ab, Vt, nullptr, outb, DMODEL, SEQ, DMODEL, 1.0f);

  // 5. vocab projection (dominant GEMM, f32 out + bias, vocab-tail guarded)
  dim3 gP((NVOCAB + 255) / 256, SEQ / 128);
  k_gemm_nt<256, 1, 1, 0><<<gP, 256, 0, stream>>>(outb, wpb, bp, logits, NVOCAB, DMODEL, NVOCAB, 1.0f);
}
